// cape_mamba_15925738733611
// MI455X (gfx1250) — compile-verified, matrix operand order checked by reading, not run on hardware
//
#include <hip/hip_runtime.h>
#include <hip/hip_bf16.h>

// ---------------------------------------------------------------- types
typedef __bf16 bf16;
typedef __attribute__((ext_vector_type(16))) __bf16 v16bf;
typedef __attribute__((ext_vector_type(8)))  __bf16 v8bf;
typedef __attribute__((ext_vector_type(8)))  float  v8f;

#define B_SZ     2
#define L_SZ     2048
#define D_MODEL  1024
#define D_INNER  2048
#define D_STATE  16
#define D_CONV   4
#define DT_RANK  64
#define ROWS     (B_SZ * L_SZ)           // 4096 token rows
#define XDBL_N   (DT_RANK + 2 * D_STATE) // 96

__device__ __forceinline__ bf16 to_bf16(float f) {
    union { float f; unsigned u; } a; a.f = f;
    unsigned r = a.u + 0x7FFFu + ((a.u >> 16) & 1u);   // round-to-nearest-even
    union { unsigned short s; bf16 b; } o; o.s = (unsigned short)(r >> 16);
    return o.b;
}

__device__ __forceinline__ float sigmoidf_(float x) { return 1.f / (1.f + __expf(-x)); }
__device__ __forceinline__ float siluf_(float x)    { return x * sigmoidf_(x); }
__device__ __forceinline__ float softplusf_(float x) {
    return fmaxf(x, 0.f) + log1pf(__expf(-fabsf(x)));
}

// ---------------------------------------------------------------- fp32 -> bf16 strided convert
__global__ __launch_bounds__(256)
void cvt2d_kernel(const float* __restrict__ src, bf16* __restrict__ dst,
                  int rows, int cols, int sld, int dld) {
    int idx = blockIdx.x * 256 + threadIdx.x;
    if (idx >= rows * cols) return;
    int r = idx / cols, c = idx - r * cols;
    dst[(size_t)r * dld + c] = to_bf16(src[(size_t)r * sld + c]);
}

// ---------------------------------------------------------------- WMMA bf16 GEMM
//   Cf[M,N] (+ optional bf16 mirror Cb) = act( A[M,K] @ W[N,K]^T + bias )
//   act: 0 = none, 1 = softplus
//   Block = 256 threads (8 waves). Wave w owns rows [blkY*128 + w*16, +16),
//   cols [colOff + blkX*64, +16*NT) as NT unconditional WMMA tiles sharing
//   one A fragment. K must be a multiple of 64 (true for all GEMMs here);
//   the K-loop is unrolled x2 with double-buffered fragments so the next
//   step's global_load_b128 clause overlaps the current step's WMMAs.
template <int NT>
__global__ __launch_bounds__(256)
void gemm_bf16_kernel(const bf16* __restrict__ A, const bf16* __restrict__ W,
                      const float* __restrict__ bias,
                      float* __restrict__ Cf, bf16* __restrict__ Cb,
                      int M, int N, int K, int act, int colOff) {
    const int lane = threadIdx.x & 31;
    const int wave = threadIdx.x >> 5;
    const int r0 = blockIdx.y * 128 + wave * 16;
    const int c0 = colOff + blockIdx.x * 64;
    if (r0 >= M) return;                       // wave-uniform: EXEC stays all-ones

    const int lrow = lane & 15;                // row/col within 16-tile
    const int sel  = lane >> 4;                // K-half selector (CDNA5 striping)
    const bf16* Arow = A + (size_t)(r0 + lrow) * K + (sel << 3);
    const bf16* Wrow[NT];
#pragma unroll
    for (int t = 0; t < NT; ++t)
        Wrow[t] = W + (size_t)(c0 + t * 16 + lrow) * K + (sel << 3);

    union Frag { v16bf v; v8bf h[2]; };
    Frag a0, a1, b0[NT], b1[NT];
    v8f acc[NT];
#pragma unroll
    for (int t = 0; t < NT; ++t) acc[t] = (v8f){};

    // prologue: fill buffer 0 with k = 0
    a0.h[0] = *(const v8bf*)(Arow + 0);
    a0.h[1] = *(const v8bf*)(Arow + 16);
#pragma unroll
    for (int t = 0; t < NT; ++t) {
        b0[t].h[0] = *(const v8bf*)(Wrow[t] + 0);
        b0[t].h[1] = *(const v8bf*)(Wrow[t] + 16);
    }

    for (int k = 0; k < K; k += 64) {
        __builtin_prefetch(Arow + k + 256, 0, 3);
        // load buffer 1 (k + 32) while consuming buffer 0 (k)
        a1.h[0] = *(const v8bf*)(Arow + k + 32);
        a1.h[1] = *(const v8bf*)(Arow + k + 48);
#pragma unroll
        for (int t = 0; t < NT; ++t) {
            b1[t].h[0] = *(const v8bf*)(Wrow[t] + k + 32);
            b1[t].h[1] = *(const v8bf*)(Wrow[t] + k + 48);
        }
#pragma unroll
        for (int t = 0; t < NT; ++t)
            acc[t] = __builtin_amdgcn_wmma_f32_16x16x32_bf16(
                false, a0.v, false, b0[t].v, (short)0, acc[t], false, false);
        // load buffer 0 (k + 64) while consuming buffer 1 (k + 32)
        if (k + 64 < K) {
            a0.h[0] = *(const v8bf*)(Arow + k + 64);
            a0.h[1] = *(const v8bf*)(Arow + k + 80);
#pragma unroll
            for (int t = 0; t < NT; ++t) {
                b0[t].h[0] = *(const v8bf*)(Wrow[t] + k + 64);
                b0[t].h[1] = *(const v8bf*)(Wrow[t] + k + 80);
            }
        }
#pragma unroll
        for (int t = 0; t < NT; ++t)
            acc[t] = __builtin_amdgcn_wmma_f32_16x16x32_bf16(
                false, a1.v, false, b1[t].v, (short)0, acc[t], false, false);
    }

    // C/D layout: element r of the v8f = row (r0 + sel*8 + r), col (tile + lrow)
    const int mrow = r0 + sel * 8;
#pragma unroll
    for (int t = 0; t < NT; ++t) {
        const int cn = c0 + t * 16 + lrow;
        const float bv = bias ? bias[cn] : 0.f;
#pragma unroll
        for (int r = 0; r < 8; ++r) {
            float v = acc[t][r] + bv;
            if (act == 1) v = softplusf_(v);
            const size_t off = (size_t)(mrow + r) * N + cn;
            if (Cf) Cf[off] = v;
            if (Cb) Cb[off] = to_bf16(v);
        }
    }
}

// ---------------------------------------------------------------- depthwise causal conv (k=4) + SiLU
__global__ __launch_bounds__(256)
void conv_silu_kernel(const float* __restrict__ xz,        // (ROWS, 2*D_INNER), xi = cols [0,D_INNER)
                      const float* __restrict__ conv_w,    // (D_INNER, 1, 4)
                      const float* __restrict__ conv_b,
                      float* __restrict__ xc_f, bf16* __restrict__ xc_b) {
    int idx = blockIdx.x * 256 + threadIdx.x;              // over B*L*D_INNER
    if (idx >= B_SZ * L_SZ * D_INNER) return;
    int d = idx & (D_INNER - 1);
    int l = (idx >> 11) & (L_SZ - 1);
    int b = idx >> 22;
    float acc = conv_b[d];
    const float w0 = conv_w[d * 4 + 0], w1 = conv_w[d * 4 + 1],
                w2 = conv_w[d * 4 + 2], w3 = conv_w[d * 4 + 3];
    const size_t rowb = (size_t)(b * L_SZ) * (2 * D_INNER);
    if (l >= 3) {
        const size_t p = rowb + (size_t)(l - 3) * (2 * D_INNER) + d;
        acc += w0 * xz[p] + w1 * xz[p + 2 * D_INNER] +
               w2 * xz[p + 4 * D_INNER] + w3 * xz[p + 6 * D_INNER];
    } else {
#pragma unroll
        for (int j = 0; j < 4; ++j) {
            int li = l - 3 + j;
            if (li >= 0) {
                float w = (j == 0) ? w0 : (j == 1) ? w1 : (j == 2) ? w2 : w3;
                acc += w * xz[rowb + (size_t)li * (2 * D_INNER) + d];
            }
        }
    }
    float s = siluf_(acc);
    xc_f[idx] = s;
    xc_b[idx] = to_bf16(s);
}

// ---------------------------------------------------------------- selective scan
//   one lane per (b, d, s); 16-lane butterfly reduce over s each step.
__global__ __launch_bounds__(256)
void scan_kernel(const float* __restrict__ delta,   // (ROWS, D_INNER)
                 const float* __restrict__ xdbl,    // (ROWS, 96): [64,80)=B, [80,96)=C
                 const float* __restrict__ xc,      // (ROWS, D_INNER) fp32
                 const float* __restrict__ xz,      // (ROWS, 2*D_INNER), z = cols [D_INNER,2*D_INNER)
                 const float* __restrict__ A_log,   // (D_INNER, 16)
                 const float* __restrict__ Dp,      // (D_INNER)
                 bf16* __restrict__ ybf) {          // (ROWS, D_INNER)
    const int tid = blockIdx.x * 256 + threadIdx.x; // 65536 lanes
    const int s = tid & 15;
    const int d = (tid >> 4) & (D_INNER - 1);
    const int b = tid >> 15;
    const float Aval = -__expf(A_log[d * D_STATE + s]);
    const float Dv = Dp[d];
    float h = 0.f;
    for (int l = 0; l < L_SZ; ++l) {
        const size_t row = (size_t)(b * L_SZ + l);
        const float dt = delta[row * D_INNER + d];
        const float u  = xc[row * D_INNER + d];
        const float Bv = xdbl[row * XDBL_N + DT_RANK + s];
        const float Cv = xdbl[row * XDBL_N + DT_RANK + D_STATE + s];
        h = __expf(dt * Aval) * h + (dt * u) * Bv;
        float p = h * Cv;
        p += __shfl_xor(p, 1);
        p += __shfl_xor(p, 2);
        p += __shfl_xor(p, 4);
        p += __shfl_xor(p, 8);
        if (s == 0) {
            const float z = xz[row * (2 * D_INNER) + D_INNER + d];
            float y = (p + u * Dv) * siluf_(z);
            ybf[row * D_INNER + d] = to_bf16(y);
        }
    }
}

// ---------------------------------------------------------------- LN + ReLU -> bf16 (detail branch hidden)
__global__ __launch_bounds__(256)
void lnrelu_kernel(const float* __restrict__ src, const float* __restrict__ g,
                   const float* __restrict__ bb, bf16* __restrict__ dst) {
    __shared__ float s_sum[256], s_sq[256];
    const int row = blockIdx.x, t = threadIdx.x;
    const size_t base = (size_t)row * D_MODEL;
    float v[4], sum = 0.f, sq = 0.f;
#pragma unroll
    for (int i = 0; i < 4; ++i) {
        float x = src[base + t + 256 * i];
        v[i] = x; sum += x; sq += x * x;
    }
    s_sum[t] = sum; s_sq[t] = sq; __syncthreads();
    for (int o = 128; o > 0; o >>= 1) {
        if (t < o) { s_sum[t] += s_sum[t + o]; s_sq[t] += s_sq[t + o]; }
        __syncthreads();
    }
    const float mean = s_sum[0] * (1.f / D_MODEL);
    const float var  = s_sq[0] * (1.f / D_MODEL) - mean * mean;
    const float inv  = rsqrtf(var + 1e-5f);
#pragma unroll
    for (int i = 0; i < 4; ++i) {
        int c = t + 256 * i;
        float y = (v[i] - mean) * inv * g[c] + bb[c];
        dst[base + c] = to_bf16(fmaxf(y, 0.f));
    }
}

// ---------------------------------------------------------------- gate mix + final LN (fp32 out)
__global__ __launch_bounds__(256)
void final_kernel(const float* __restrict__ x, const float* __restrict__ gate_pre,
                  const float* __restrict__ m, const float* __restrict__ detail,
                  const float* __restrict__ norm_g, const float* __restrict__ norm_b,
                  const float* __restrict__ bal, float* __restrict__ out) {
    __shared__ float s_sum[256], s_sq[256];
    const int row = blockIdx.x, t = threadIdx.x;
    const size_t base = (size_t)row * D_MODEL;
    const float f = sigmoidf_(bal[0]);
    float v[4], sum = 0.f, sq = 0.f;
#pragma unroll
    for (int i = 0; i < 4; ++i) {
        int c = t + 256 * i;
        float xv = x[base + c];
        float g = sigmoidf_(gate_pre[base + c]);
        float cv = xv * g + (m[base + c] * f + detail[base + c] * (1.f - f)) * (1.f - g);
        v[i] = cv; sum += cv; sq += cv * cv;
    }
    s_sum[t] = sum; s_sq[t] = sq; __syncthreads();
    for (int o = 128; o > 0; o >>= 1) {
        if (t < o) { s_sum[t] += s_sum[t + o]; s_sq[t] += s_sq[t + o]; }
        __syncthreads();
    }
    const float mean = s_sum[0] * (1.f / D_MODEL);
    const float var  = s_sq[0] * (1.f / D_MODEL) - mean * mean;
    const float inv  = rsqrtf(var + 1e-5f);
#pragma unroll
    for (int i = 0; i < 4; ++i) {
        int c = t + 256 * i;
        out[base + c] = (v[i] - mean) * inv * norm_g[c] + norm_b[c];
    }
}

// ---------------------------------------------------------------- host
static inline int cdiv(int a, int b) { return (a + b - 1) / b; }

extern "C" void kernel_launch(void* const* d_in, const int* in_sizes, int n_in,
                              void* d_out, int out_size, void* d_ws, size_t ws_size,
                              hipStream_t stream) {
    const float* x         = (const float*)d_in[0];
    const float* in_proj_w = (const float*)d_in[1];
    const float* conv_w    = (const float*)d_in[2];
    const float* conv_b    = (const float*)d_in[3];
    const float* x_proj_w  = (const float*)d_in[4];
    const float* dt_proj_w = (const float*)d_in[5];
    const float* dt_proj_b = (const float*)d_in[6];
    const float* A_log     = (const float*)d_in[7];
    const float* Dp        = (const float*)d_in[8];
    const float* out_proj_w= (const float*)d_in[9];
    const float* gate_w    = (const float*)d_in[10];
    const float* gate_b    = (const float*)d_in[11];
    const float* fd_w1     = (const float*)d_in[12];
    const float* fd_b1     = (const float*)d_in[13];
    const float* fd_ln_g   = (const float*)d_in[14];
    const float* fd_ln_b   = (const float*)d_in[15];
    const float* fd_w2     = (const float*)d_in[16];
    const float* fd_b2     = (const float*)d_in[17];
    const float* norm_g    = (const float*)d_in[18];
    const float* norm_b    = (const float*)d_in[19];
    const float* balance   = (const float*)d_in[20];
    float* out = (float*)d_out;

    // ---- workspace carve-out (256-B aligned)
    char* ws = (char*)d_ws;
    size_t off = 0;
    auto carve = [&](size_t bytes) -> void* {
        void* p = ws + off;
        off += (bytes + 255) & ~(size_t)255;
        return p;
    };
    bf16* w_in_bf   = (bf16*)carve((size_t)2 * D_INNER * D_MODEL * 2);   // 4096x1024
    bf16* w_xp_bf   = (bf16*)carve((size_t)XDBL_N * D_INNER * 2);        //   96x2048
    bf16* w_dt_bf   = (bf16*)carve((size_t)D_INNER * DT_RANK * 2);       // 2048x64
    bf16* w_out_bf  = (bf16*)carve((size_t)D_MODEL * D_INNER * 2);       // 1024x2048
    bf16* w_gate_bf = (bf16*)carve((size_t)D_MODEL * D_MODEL * 2);
    bf16* w_fd1_bf  = (bf16*)carve((size_t)D_MODEL * D_MODEL * 2);
    bf16* w_fd2_bf  = (bf16*)carve((size_t)D_MODEL * D_MODEL * 2);
    bf16* x_bf      = (bf16*)carve((size_t)ROWS * D_MODEL * 2);
    float* xz       = (float*)carve((size_t)ROWS * 2 * D_INNER * 4);     // 64 MB
    float* gate_pre = (float*)carve((size_t)ROWS * D_MODEL * 4);
    float* h1_pre   = (float*)carve((size_t)ROWS * D_MODEL * 4);
    bf16*  h1_bf    = (bf16*)carve((size_t)ROWS * D_MODEL * 2);
    float* detail   = (float*)carve((size_t)ROWS * D_MODEL * 4);
    float* xc_f     = (float*)carve((size_t)ROWS * D_INNER * 4);
    bf16*  xc_bf    = (bf16*)carve((size_t)ROWS * D_INNER * 2);
    float* xdbl     = (float*)carve((size_t)ROWS * XDBL_N * 4);
    bf16*  dt_bf    = (bf16*)carve((size_t)ROWS * DT_RANK * 2);
    float* delta    = (float*)carve((size_t)ROWS * D_INNER * 4);
    bf16*  y_bf     = (bf16*)carve((size_t)ROWS * D_INNER * 2);
    float* m_buf    = (float*)carve((size_t)ROWS * D_MODEL * 4);
    (void)ws_size; (void)in_sizes; (void)n_in; (void)out_size;

    auto cvt = [&](const float* s, bf16* d, int rows, int cols, int sld, int dld) {
        cvt2d_kernel<<<cdiv(rows * cols, 256), 256, 0, stream>>>(s, d, rows, cols, sld, dld);
    };
    // N%16 == 0 always; full 64-col blocks use NT=4, remainder gets its own launch.
    auto gemm = [&](const bf16* A, const bf16* W, const float* bias,
                    float* Cf, bf16* Cb, int M, int N, int K, int act) {
        const int full = N / 64;
        const int rem  = (N % 64) / 16;
        const int gy = cdiv(M, 128);
        if (full > 0)
            gemm_bf16_kernel<4><<<dim3(full, gy), 256, 0, stream>>>(
                A, W, bias, Cf, Cb, M, N, K, act, 0);
        if (rem == 1)
            gemm_bf16_kernel<1><<<dim3(1, gy), 256, 0, stream>>>(
                A, W, bias, Cf, Cb, M, N, K, act, full * 64);
        else if (rem == 2)
            gemm_bf16_kernel<2><<<dim3(1, gy), 256, 0, stream>>>(
                A, W, bias, Cf, Cb, M, N, K, act, full * 64);
        else if (rem == 3)
            gemm_bf16_kernel<3><<<dim3(1, gy), 256, 0, stream>>>(
                A, W, bias, Cf, Cb, M, N, K, act, full * 64);
    };

    // 1) bf16 conversions (weights + x)
    cvt(in_proj_w, w_in_bf,   2 * D_INNER, D_MODEL, D_MODEL, D_MODEL);
    cvt(x_proj_w,  w_xp_bf,   XDBL_N,      D_INNER, D_INNER, D_INNER);
    cvt(dt_proj_w, w_dt_bf,   D_INNER,     DT_RANK, DT_RANK, DT_RANK);
    cvt(out_proj_w,w_out_bf,  D_MODEL,     D_INNER, D_INNER, D_INNER);
    cvt(gate_w,    w_gate_bf, D_MODEL,     D_MODEL, D_MODEL, D_MODEL);
    cvt(fd_w1,     w_fd1_bf,  D_MODEL,     D_MODEL, D_MODEL, D_MODEL);
    cvt(fd_w2,     w_fd2_bf,  D_MODEL,     D_MODEL, D_MODEL, D_MODEL);
    cvt(x,         x_bf,      ROWS,        D_MODEL, D_MODEL, D_MODEL);

    // 2) xz = x @ in_proj_w^T         (ROWS x 4096)
    gemm(x_bf, w_in_bf, nullptr, xz, nullptr, ROWS, 2 * D_INNER, D_MODEL, 0);
    // 3) gate_pre = x @ gate_w^T + gate_b
    gemm(x_bf, w_gate_bf, gate_b, gate_pre, nullptr, ROWS, D_MODEL, D_MODEL, 0);
    // 4) detail branch: h1 = relu(ln(x @ fd_w1^T + fd_b1)); detail = h1 @ fd_w2^T + fd_b2
    gemm(x_bf, w_fd1_bf, fd_b1, h1_pre, nullptr, ROWS, D_MODEL, D_MODEL, 0);
    lnrelu_kernel<<<ROWS, 256, 0, stream>>>(h1_pre, fd_ln_g, fd_ln_b, h1_bf);
    gemm(h1_bf, w_fd2_bf, fd_b2, detail, nullptr, ROWS, D_MODEL, D_MODEL, 0);
    // 5) xc = silu(depthwise_conv(xi) + conv_b)
    conv_silu_kernel<<<cdiv(ROWS * D_INNER, 256), 256, 0, stream>>>(
        xz, conv_w, conv_b, xc_f, xc_bf);
    // 6) x_dbl = xc @ x_proj_w^T      (ROWS x 96): 64-col NT=4 launch + 32-col NT=2 launch
    gemm(xc_bf, w_xp_bf, nullptr, xdbl, nullptr, ROWS, XDBL_N, D_INNER, 0);
    // 7) delta = softplus(dt @ dt_proj_w^T + dt_proj_b)
    cvt(xdbl, dt_bf, ROWS, DT_RANK, XDBL_N, DT_RANK);
    gemm(dt_bf, w_dt_bf, dt_proj_b, delta, nullptr, ROWS, D_INNER, DT_RANK, 1);
    // 8) selective scan + D-path + silu(z) gating -> y (bf16); 65536 lanes
    scan_kernel<<<256, 256, 0, stream>>>(delta, xdbl, xc_f, xz, A_log, Dp, y_bf);
    // 9) m = y @ out_proj_w^T
    gemm(y_bf, w_out_bf, nullptr, m_buf, nullptr, ROWS, D_MODEL, D_INNER, 0);
    // 10) gate mix + final layernorm
    final_kernel<<<ROWS, 256, 0, stream>>>(x, gate_pre, m_buf, detail,
                                           norm_g, norm_b, balance, out);
}
